// PCAM_Module_13314398618076
// MI455X (gfx1250) — 16-bit and 8-bit matrix operand order repaired and checked by reading, compiled, not run on hardware
//
#include <hip/hip_runtime.h>


typedef __attribute__((ext_vector_type(16))) __bf16 v16bf;
typedef __attribute__((ext_vector_type(16))) int    v16i;
typedef __attribute__((ext_vector_type(8)))  float  v8f;

#define BATCH 2
#define CIN   2048
#define NPIX  4096
#define C4    512
#define C16   128
#define HH    64
#define WW2   64
#define PW    66
#define NPAD  (PW * PW)

struct U16x16 { uint4 lo, hi; };
struct U8x64A { uint2 c[8]; };
struct U8x64B { uint4 c[4]; };

__device__ __forceinline__ unsigned short f2bf(float f) {
    unsigned int u = __builtin_bit_cast(unsigned int, f);
    u += 0x7FFFu + ((u >> 16) & 1u);
    return (unsigned short)(u >> 16);
}

__device__ __forceinline__ unsigned char f2fp8(float f) {
    unsigned int u = __builtin_bit_cast(unsigned int, f);
    unsigned int sgn = (u >> 24) & 0x80u;
    unsigned int au = u & 0x7FFFFFFFu;
    if (au >= 0x43E00000u) return (unsigned char)(sgn | 0x7Eu);
    if (au < 0x3C800000u) {
        float a = __builtin_bit_cast(float, au);
        int q = (int)(a * 512.0f + 0.5f);
        return (unsigned char)(sgn | (unsigned)q);
    }
    unsigned int m = au + 0x7FFFFu + ((au >> 20) & 1u);
    int ee = (int)((m >> 23) & 0xFFu) - 120;
    if (ee >= 16) return (unsigned char)(sgn | 0x7Eu);
    return (unsigned char)(sgn | ((unsigned)ee << 3) | ((m >> 20) & 0x7u));
}

__device__ __forceinline__ v8f wmma_bf16(v16bf a, v16bf b, v8f c) {
    return __builtin_amdgcn_wmma_f32_16x16x32_bf16(false, a, false, b,
                                                   (short)0, c, false, false);
}
__device__ __forceinline__ v8f wmma_fp8(v16i a, v16i b, v8f c) {
    return __builtin_amdgcn_wmma_f32_16x16x128_fp8_fp8(a, b, (short)0, c,
                                                       false, false);
}

__device__ __forceinline__ v16bf ld_afrag(const unsigned short* sA, int row0, int lane) {
    const int lm = row0 + (lane & 15);
    const int kb = (lane < 16) ? 0 : 8;
    U16x16 p;
    p.lo = *(const uint4*)(sA + lm * 40 + kb);
    p.hi = *(const uint4*)(sA + lm * 40 + 16 + kb);
    return __builtin_bit_cast(v16bf, p);
}
__device__ __forceinline__ v16bf ld_bfrag(const unsigned short* sB, int col0, int lane) {
    const int n = col0 + (lane & 15);
    const int kq = (lane < 16) ? 0 : 8;
    U16x16 p;
    p.lo = *(const uint4*)(sB + n * 40 + kq);
    p.hi = *(const uint4*)(sB + n * 40 + 16 + kq);
    return __builtin_bit_cast(v16bf, p);
}

__device__ __forceinline__ v16i ld_afrag8(const unsigned char* sA, int row0, int lane) {
    const unsigned char* r = sA + (row0 + (lane & 15)) * 144 + ((lane < 16) ? 0 : 16);
    U8x64A p;
#pragma unroll
    for (int j = 0; j < 8; ++j) p.c[j] = *(const uint2*)(r + 32 * (j >> 1) + 8 * (j & 1));
    return __builtin_bit_cast(v16i, p);
}
__device__ __forceinline__ v16i ld_bfrag8(const unsigned char* sB, int col0, int lane) {
    const unsigned char* r = sB + (col0 + (lane & 15)) * 144 + ((lane < 16) ? 0 : 16);
    U8x64B p;
#pragma unroll
    for (int j = 0; j < 4; ++j) p.c[j] = *(const uint4*)(r + 32 * j);
    return __builtin_bit_cast(v16i, p);
}

template <int EPI, int ONF, int OTF, int OPAD>
__global__ __launch_bounds__(256) void gemm_tn(
    const unsigned short* __restrict__ A, int lda,
    const unsigned short* __restrict__ Bm, int ldb,
    void* __restrict__ OutN, int ldn,
    void* __restrict__ OutT, int ldt,
    int M, int Ncols, int K,
    const float* __restrict__ bias,
    const float* __restrict__ scl,
    const float* __restrict__ sh,
    const float* __restrict__ other,
    const float* __restrict__ gscal) {
    __shared__ __align__(16) unsigned short sA[2][128 * 40];
    __shared__ __align__(16) unsigned short sB[2][128 * 40];
    const int tid = threadIdx.x, lane = tid & 31, wv = tid >> 5;
    const int wm = wv >> 1, wn = wv & 1;
    const int m0 = blockIdx.y * 128, n0 = blockIdx.x * 128;

    v8f acc[2][4] = {};

    const int frow = tid >> 1;
    const int fk   = (tid & 1) * 16;
    const unsigned short* arow = A  + (size_t)(m0 + frow) * lda + fk;
    const unsigned short* brow = Bm + (size_t)(n0 + frow) * ldb + fk;
    const int sOff = frow * 40 + fk;
    const int nk = K >> 5;

    {
        uint4 a0 = *(const uint4*)(arow);
        uint4 a1 = *(const uint4*)(arow + 8);
        uint4 b0 = *(const uint4*)(brow);
        uint4 b1 = *(const uint4*)(brow + 8);
        *(uint4*)(&sA[0][0] + sOff)     = a0;
        *(uint4*)(&sA[0][0] + sOff + 8) = a1;
        *(uint4*)(&sB[0][0] + sOff)     = b0;
        *(uint4*)(&sB[0][0] + sOff + 8) = b1;
    }

    const unsigned short* anext = arow + 32;
    const unsigned short* bnext = brow + 32;
    const unsigned short* apf   = arow + 64;
    const unsigned short* bpf   = brow + 64;

    for (int i = 0; i < nk; ++i) {
        __syncthreads();
        const int cur = i & 1;
        const bool more = (i + 1) < nk;
        uint4 na0, na1, nb0, nb1;
        if (more) {
            na0 = *(const uint4*)(anext);
            na1 = *(const uint4*)(anext + 8);
            nb0 = *(const uint4*)(bnext);
            nb1 = *(const uint4*)(bnext + 8);
            anext += 32; bnext += 32;
        }
        if (i + 2 < nk) {
            __builtin_prefetch(apf, 0, 1);
            __builtin_prefetch(bpf, 0, 1);
            apf += 32; bpf += 32;
        }
        const unsigned short* cA = &sA[cur][0];
        const unsigned short* cB = &sB[cur][0];
        v16bf af0 = ld_afrag(cA, wm * 32, lane);
        v16bf af1 = ld_afrag(cA, wm * 32 + 16, lane);
#pragma unroll
        for (int s = 0; s < 4; ++s) {
            v16bf bf = ld_bfrag(cB, wn * 64 + s * 16, lane);
            acc[0][s] = wmma_bf16(af0, bf, acc[0][s]);
            acc[1][s] = wmma_bf16(af1, bf, acc[1][s]);
        }
        if (more) {
            const int nxt = cur ^ 1;
            *(uint4*)(&sA[nxt][0] + sOff)     = na0;
            *(uint4*)(&sA[nxt][0] + sOff + 8) = na1;
            *(uint4*)(&sB[nxt][0] + sOff)     = nb0;
            *(uint4*)(&sB[nxt][0] + sOff + 8) = nb1;
        }
    }

    const int ln = lane & 15;
    const float g = (EPI == 3) ? gscal[0] : 0.f;
#pragma unroll
    for (int a2 = 0; a2 < 2; ++a2) {
        const int mrb = m0 + wm * 32 + a2 * 16 + ((lane < 16) ? 0 : 8);
#pragma unroll
        for (int s = 0; s < 4; ++s) {
            const int nc = n0 + wn * 64 + s * 16 + ln;
            float v[8];
#pragma unroll
            for (int e = 0; e < 8; ++e) {
                float val = acc[a2][s][e];
                if (EPI >= 1) val += bias[mrb + e];
                if (EPI >= 2) {
                    val = scl[mrb + e] * val + sh[mrb + e];
                    val = val > 0.f ? val : 0.f;
                }
                v[e] = val;
            }
            if (EPI == 3) {
                const float* op = other + (size_t)nc * ldt + mrb;
                float4 o0 = *(const float4*)(op);
                float4 o1 = *(const float4*)(op + 4);
                v[0] += g * o0.x; v[1] += g * o0.y; v[2] += g * o0.z; v[3] += g * o0.w;
                v[4] += g * o1.x; v[5] += g * o1.y; v[6] += g * o1.z; v[7] += g * o1.w;
            }
            if (ONF == 1) {
#pragma unroll
                for (int e = 0; e < 8; ++e)
                    ((float*)OutN)[(size_t)(mrb + e) * ldn + nc] = v[e];
            } else if (ONF == 2) {
#pragma unroll
                for (int e = 0; e < 8; ++e)
                    ((unsigned short*)OutN)[(size_t)(mrb + e) * ldn + nc] = f2bf(v[e]);
            }
            const int np = OPAD ? (((nc >> 6) + 1) * PW + (nc & 63) + 1) : nc;
            if (OTF == 1) {
                float* o = (float*)OutT + (size_t)np * ldt + mrb;
                *(float4*)(o)     = make_float4(v[0], v[1], v[2], v[3]);
                *(float4*)(o + 4) = make_float4(v[4], v[5], v[6], v[7]);
            } else if (OTF == 2) {
                uint4 pk;
                pk.x = (unsigned)f2bf(v[0]) | ((unsigned)f2bf(v[1]) << 16);
                pk.y = (unsigned)f2bf(v[2]) | ((unsigned)f2bf(v[3]) << 16);
                pk.z = (unsigned)f2bf(v[4]) | ((unsigned)f2bf(v[5]) << 16);
                pk.w = (unsigned)f2bf(v[6]) | ((unsigned)f2bf(v[7]) << 16);
                *(uint4*)((unsigned short*)OutT + (size_t)np * ldt + mrb) = pk;
            }
        }
    }
}

__global__ __launch_bounds__(256) void gemm_fp8_pv(
    const unsigned char* __restrict__ P,
    const unsigned char* __restrict__ V,
    const float* __restrict__ rscale,
    float* __restrict__ Out, int ldn) {
    __shared__ __align__(16) unsigned char sA[2][128 * 144];
    __shared__ __align__(16) unsigned char sB[2][128 * 144];
    const int tid = threadIdx.x, lane = tid & 31, wv = tid >> 5;
    const int wm = wv >> 1, wn = wv & 1;
    const int m0 = blockIdx.y * 128, n0 = blockIdx.x * 128;

    v8f acc[2][4] = {};

    const int frow = tid >> 1;
    const int fh   = (tid & 1) * 64;
    const unsigned char* arow = P + (size_t)(m0 + frow) * NPIX + fh;
    const unsigned char* brow = V + (size_t)(n0 + frow) * NPIX + fh;
    const int sOff = frow * 144 + fh;
    const int nk = NPIX / 128;

    {
#pragma unroll
        for (int j = 0; j < 4; ++j) {
            *(uint4*)(&sA[0][0] + sOff + 16 * j) = *(const uint4*)(arow + 16 * j);
            *(uint4*)(&sB[0][0] + sOff + 16 * j) = *(const uint4*)(brow + 16 * j);
        }
    }

    const unsigned char* anext = arow + 128;
    const unsigned char* bnext = brow + 128;
    const unsigned char* apf   = arow + 256;
    const unsigned char* bpf   = brow + 256;

    for (int i = 0; i < nk; ++i) {
        __syncthreads();
        const int cur = i & 1;
        const bool more = (i + 1) < nk;
        uint4 na[4], nb[4];
        if (more) {
#pragma unroll
            for (int j = 0; j < 4; ++j) {
                na[j] = *(const uint4*)(anext + 16 * j);
                nb[j] = *(const uint4*)(bnext + 16 * j);
            }
            anext += 128; bnext += 128;
        }
        if (i + 2 < nk) {
            __builtin_prefetch(apf, 0, 1);
            __builtin_prefetch(bpf, 0, 1);
            apf += 128; bpf += 128;
        }
        const unsigned char* cA = &sA[cur][0];
        const unsigned char* cB = &sB[cur][0];
        v16i af0 = ld_afrag8(cA, wm * 32, lane);
        v16i af1 = ld_afrag8(cA, wm * 32 + 16, lane);
#pragma unroll
        for (int s = 0; s < 4; ++s) {
            v16i bf = ld_bfrag8(cB, wn * 64 + s * 16, lane);
            acc[0][s] = wmma_fp8(af0, bf, acc[0][s]);
            acc[1][s] = wmma_fp8(af1, bf, acc[1][s]);
        }
        if (more) {
            const int nxt = cur ^ 1;
#pragma unroll
            for (int j = 0; j < 4; ++j) {
                *(uint4*)(&sA[nxt][0] + sOff + 16 * j) = na[j];
                *(uint4*)(&sB[nxt][0] + sOff + 16 * j) = nb[j];
            }
        }
    }

    const int ln = lane & 15;
#pragma unroll
    for (int a2 = 0; a2 < 2; ++a2) {
        const int mrb = m0 + wm * 32 + a2 * 16 + ((lane < 16) ? 0 : 8);
#pragma unroll
        for (int s = 0; s < 4; ++s) {
            const int nc = n0 + wn * 64 + s * 16 + ln;
#pragma unroll
            for (int e = 0; e < 8; ++e)
                Out[(size_t)(mrb + e) * ldn + nc] = rscale[mrb + e] * acc[a2][s][e];
        }
    }
}

__global__ __launch_bounds__(256) void conv3x3_wmma(
    const unsigned short* __restrict__ inP,
    const unsigned short* __restrict__ wpk,
    const float* __restrict__ bias,
    const float* __restrict__ scl,
    const float* __restrict__ sh,
    float* __restrict__ outF,
    unsigned short* __restrict__ outT) {
    __shared__ __align__(16) unsigned short sA[2][64 * 40];
    __shared__ __align__(16) unsigned short sB[2][64 * 40];
    const int tid = threadIdx.x, lane = tid & 31, wv = tid >> 5;
    const int wm = wv >> 1, wn = wv & 1;
    const int h = blockIdx.x;
    const int oc0 = blockIdx.y * 64;
    const int N = HH * WW2;
    const int NSTEP = 9 * (C4 / 32);

    v8f acc0 = {}, acc1 = {};
    const int amm = tid >> 2, akk = (tid & 3) * 8;
    const int sOff = amm * 40 + akk;

    const unsigned short* abase = wpk + (size_t)(oc0 + amm) * C4 + akk;
    const unsigned short* bbase = inP + ((size_t)(h + 1) * PW + amm + 1) * C4 + akk;

    auto fetch = [&](int it, uint4& av, uint4& bvv) {
        const int tap = it >> 4;
        const int kc  = (it & 15) << 5;
        const int q   = (tap * 11) >> 5;
        const int bdel = (63 * q + tap - 67) * C4;
        av  = *(const uint4*)(abase + ((size_t)tap * C4 * C4 + kc));
        bvv = *(const uint4*)(bbase + (bdel + kc));
    };

    {
        uint4 av, bvv;
        fetch(0, av, bvv);
        *(uint4*)(&sA[0][0] + sOff) = av;
        *(uint4*)(&sB[0][0] + sOff) = bvv;
    }

    for (int i = 0; i < NSTEP; ++i) {
        __syncthreads();
        const int cur = i & 1;
        const bool more = (i + 1) < NSTEP;
        uint4 nav, nbv;
        if (more) fetch(i + 1, nav, nbv);
        const unsigned short* cA = &sA[cur][0];
        const unsigned short* cB = &sB[cur][0];
        v16bf af = ld_afrag(cA, wm * 16, lane);
        v16bf b0 = ld_bfrag(cB, wn * 32, lane);
        v16bf b1 = ld_bfrag(cB, wn * 32 + 16, lane);
        acc0 = wmma_bf16(af, b0, acc0);
        acc1 = wmma_bf16(af, b1, acc1);
        if (more) {
            const int nxt = cur ^ 1;
            *(uint4*)(&sA[nxt][0] + sOff) = nav;
            *(uint4*)(&sB[nxt][0] + sOff) = nbv;
        }
    }

    const int ln  = lane & 15;
    const int ocb = oc0 + wm * 16 + ((lane < 16) ? 0 : 8);
#pragma unroll
    for (int t = 0; t < 2; ++t) {
        v8f a = t ? acc1 : acc0;
        const int p = h * WW2 + wn * 32 + t * 16 + ln;
        float v[8];
#pragma unroll
        for (int e = 0; e < 8; ++e) {
            float val = a[e] + bias[ocb + e];
            val = scl[ocb + e] * val + sh[ocb + e];
            v[e] = val > 0.f ? val : 0.f;
        }
#pragma unroll
        for (int e = 0; e < 8; ++e)
            outF[(size_t)(ocb + e) * N + p] = v[e];
        if (outT) {
            uint4 pk;
            pk.x = (unsigned)f2bf(v[0]) | ((unsigned)f2bf(v[1]) << 16);
            pk.y = (unsigned)f2bf(v[2]) | ((unsigned)f2bf(v[3]) << 16);
            pk.z = (unsigned)f2bf(v[4]) | ((unsigned)f2bf(v[5]) << 16);
            pk.w = (unsigned)f2bf(v[6]) | ((unsigned)f2bf(v[7]) << 16);
            *(uint4*)(outT + (size_t)p * C4 + ocb) = pk;
        }
    }
}

__global__ __launch_bounds__(256) void softmax_row_fp8(float* __restrict__ S,
                                                       unsigned char* __restrict__ P8,
                                                       float* __restrict__ rsum,
                                                       int N) {
    __shared__ float red[8];
    __shared__ float bval;
    const size_t base = (size_t)blockIdx.x * N;
    float* s = S + base;
    const int tid = threadIdx.x, lane = tid & 31, w = tid >> 5;

    float mx = -3.4e38f;
    for (int i = tid; i < N; i += 256) mx = fmaxf(mx, s[i]);
    for (int o = 16; o > 0; o >>= 1) mx = fmaxf(mx, __shfl_xor(mx, o, 32));
    if (lane == 0) red[w] = mx;
    __syncthreads();
    if (tid == 0) { float m = red[0]; for (int i = 1; i < 8; ++i) m = fmaxf(m, red[i]); bval = m; }
    __syncthreads();
    mx = bval;

    float sum = 0.f;
    for (int i = tid; i < N; i += 256) {
        float e = __expf(s[i] - mx);
        P8[base + i] = f2fp8(e);
        sum += e;
    }
    for (int o = 16; o > 0; o >>= 1) sum += __shfl_xor(sum, o, 32);
    if (lane == 0) red[w] = sum;
    __syncthreads();
    if (tid == 0) {
        float m = 0.f;
        for (int i = 0; i < 8; ++i) m += red[i];
        rsum[blockIdx.x] = 1.f / m;
    }
}

__global__ __launch_bounds__(256) void softmax_chan(const float* __restrict__ E,
                                                    unsigned short* __restrict__ A,
                                                    int N) {
    __shared__ float buf[512];
    __shared__ float red[8];
    __shared__ float bval;
    const size_t base = (size_t)blockIdx.x * N;
    const int tid = threadIdx.x, lane = tid & 31, w = tid >> 5;

    float mn = 3.4e38f;
    for (int i = tid; i < N; i += 256) { float v = E[base + i]; buf[i] = v; mn = fminf(mn, v); }
    for (int o = 16; o > 0; o >>= 1) mn = fminf(mn, __shfl_xor(mn, o, 32));
    if (lane == 0) red[w] = mn;
    __syncthreads();
    if (tid == 0) { float m = red[0]; for (int i = 1; i < 8; ++i) m = fminf(m, red[i]); bval = m; }
    __syncthreads();
    mn = bval;

    float sum = 0.f;
    for (int i = tid; i < N; i += 256) { float p = __expf(mn - buf[i]); buf[i] = p; sum += p; }
    for (int o = 16; o > 0; o >>= 1) sum += __shfl_xor(sum, o, 32);
    if (lane == 0) red[w] = sum;
    __syncthreads();
    if (tid == 0) { float m = 0.f; for (int i = 0; i < 8; ++i) m += red[i]; bval = m; }
    __syncthreads();
    const float r = 1.f / bval;
    for (int i = tid; i < N; i += 256) A[base + i] = f2bf(buf[i] * r);
}

__global__ __launch_bounds__(256) void cast_bf16(const float* __restrict__ in,
                                                 unsigned short* __restrict__ out,
                                                 int n) {
    int i = blockIdx.x * 256 + threadIdx.x;
    if (i < n) out[i] = f2bf(in[i]);
}

__global__ __launch_bounds__(256) void zero_u16(unsigned short* __restrict__ p, int n) {
    int i = blockIdx.x * 256 + threadIdx.x;
    if (i < n) p[i] = 0;
}

__global__ __launch_bounds__(256) void bf16_to_fp8(const unsigned short* __restrict__ in,
                                                   unsigned char* __restrict__ out,
                                                   int n) {
    int i = blockIdx.x * 256 + threadIdx.x;
    if (i < n) {
        float f = __builtin_bit_cast(float, (unsigned int)in[i] << 16);
        out[i] = f2fp8(f);
    }
}

__global__ __launch_bounds__(256) void transpose_cast(const float* __restrict__ in, int ldin,
                                                      unsigned short* __restrict__ out, int ldout) {
    __shared__ unsigned short t[64 * 65];
    const int r0 = blockIdx.y * 64, c0 = blockIdx.x * 64;
    const int tid = threadIdx.x;
#pragma unroll
    for (int j = 0; j < 16; ++j) {
        int e = tid + 256 * j;
        int rr = e >> 6, cc = e & 63;
        t[rr * 65 + cc] = f2bf(in[(size_t)(r0 + rr) * ldin + (c0 + cc)]);
    }
    __syncthreads();
#pragma unroll
    for (int j = 0; j < 16; ++j) {
        int e = tid + 256 * j;
        int cc = e >> 6, rr = e & 63;
        out[(size_t)(c0 + cc) * ldout + (r0 + rr)] = t[rr * 65 + cc];
    }
}

__global__ __launch_bounds__(256) void pack_conv_w(const float* __restrict__ w,
                                                   unsigned short* __restrict__ out,
                                                   int cio) {
    int i = blockIdx.x * 256 + threadIdx.x;
    if (i < 9 * cio) {
        int tap = i / cio, oi = i - tap * cio;
        out[i] = f2bf(w[(size_t)oi * 9 + tap]);
    }
}

extern "C" void kernel_launch(void* const* d_in, const int* in_sizes, int n_in,
                              void* d_out, int out_size, void* d_ws, size_t ws_size,
                              hipStream_t stream) {
    (void)in_sizes; (void)n_in; (void)out_size; (void)ws_size;
    const float* x     = (const float*)d_in[0];
    const float* wq_p  = (const float*)d_in[1];  const float* bq_p = (const float*)d_in[2];
    const float* wpc   = (const float*)d_in[3];  const float* bpc  = (const float*)d_in[4];
    const float* spc   = (const float*)d_in[5];  const float* tpc  = (const float*)d_in[6];
    const float* wk    = (const float*)d_in[7];  const float* bk   = (const float*)d_in[8];
    const float* wv    = (const float*)d_in[9];  const float* bv   = (const float*)d_in[10];
    const float* wrp   = (const float*)d_in[11]; const float* brp  = (const float*)d_in[12];
    const float* srp   = (const float*)d_in[13]; const float* trp  = (const float*)d_in[14];
    const float* wpf   = (const float*)d_in[15]; const float* bpf  = (const float*)d_in[16];
    const float* spf   = (const float*)d_in[17]; const float* tpf  = (const float*)d_in[18];
    const float* gam_p = (const float*)d_in[19];
    const float* wq_c  = (const float*)d_in[20]; const float* bq_c = (const float*)d_in[21];
    const float* wrc   = (const float*)d_in[22]; const float* brc  = (const float*)d_in[23];
    const float* src_  = (const float*)d_in[24]; const float* trc  = (const float*)d_in[25];
    const float* wcf   = (const float*)d_in[26]; const float* bcf  = (const float*)d_in[27];
    const float* scf   = (const float*)d_in[28]; const float* tcf  = (const float*)d_in[29];
    const float* gam_c = (const float*)d_in[30];

    float* out_c = (float*)d_out;
    float* out_p = (float*)d_out + (size_t)BATCH * C4 * NPIX;

    char* wsp = (char*)d_ws;
    auto take = [&](size_t bytes) -> void* {
        void* p = wsp; wsp += (bytes + 255) & ~(size_t)255; return p;
    };
    unsigned short* wqpb  = (unsigned short*)take((size_t)C16 * CIN * 2);
    unsigned short* wpcb  = (unsigned short*)take((size_t)C4 * CIN * 2);
    unsigned short* wkb   = (unsigned short*)take((size_t)C16 * C4 * 2);
    unsigned short* wvb   = (unsigned short*)take((size_t)C4 * CIN * 2);
    unsigned short* wrpb  = (unsigned short*)take((size_t)C4 * CIN * 2);
    unsigned short* wqcb  = (unsigned short*)take((size_t)C4 * CIN * 2);
    unsigned short* wrcb  = (unsigned short*)take((size_t)C4 * CIN * 2);
    unsigned short* wpfk  = (unsigned short*)take((size_t)9 * C4 * C4 * 2);
    unsigned short* wcfk  = (unsigned short*)take((size_t)9 * C4 * C4 * 2);
    unsigned short* xbt   = (unsigned short*)take((size_t)NPIX * CIN * 2);
    unsigned short* qpt   = (unsigned short*)take((size_t)NPIX * C16 * 2);
    unsigned short* pcb   = (unsigned short*)take((size_t)C4 * NPIX * 2);
    unsigned short* pct   = (unsigned short*)take((size_t)NPIX * C4 * 2);
    unsigned short* kpt   = (unsigned short*)take((size_t)NPIX * C16 * 2);
    unsigned short* vpb   = (unsigned short*)take((size_t)C4 * NPIX * 2);
    unsigned char*  vp8   = (unsigned char*)take((size_t)C4 * NPIX);
    unsigned short* qcb   = (unsigned short*)take((size_t)C4 * NPIX * 2);
    float*          Smat  = (float*)take((size_t)NPIX * NPIX * 4);
    unsigned char*  Pb8   = (unsigned char*)take((size_t)NPIX * NPIX);
    float*          rsum  = (float*)take((size_t)NPIX * 4);
    float*          Of    = (float*)take((size_t)NPIX * C4 * 4);
    unsigned short* ppad  = (unsigned short*)take((size_t)NPAD * C4 * 2);
    unsigned short* popt  = (unsigned short*)take((size_t)NPIX * C4 * 2);
    float*          Emat  = (float*)take((size_t)C4 * C4 * 4);
    unsigned short* acb   = (unsigned short*)take((size_t)C4 * C4 * 2);

    zero_u16<<<dim3((NPAD * C4 + 255) / 256), 256, 0, stream>>>(ppad, NPAD * C4);

    auto cast = [&](const float* s, unsigned short* d, size_t n) {
        cast_bf16<<<dim3((unsigned)((n + 255) / 256)), 256, 0, stream>>>(s, d, (int)n);
    };
    cast(wq_p, wqpb, (size_t)C16 * CIN);
    cast(wpc, wpcb, (size_t)C4 * CIN);
    cast(wk, wkb, (size_t)C16 * C4);
    cast(wv, wvb, (size_t)C4 * CIN);
    cast(wrp, wrpb, (size_t)C4 * CIN);
    cast(wq_c, wqcb, (size_t)C4 * CIN);
    cast(wrc, wrcb, (size_t)C4 * CIN);
    pack_conv_w<<<dim3((9 * C4 * C4) / 256), 256, 0, stream>>>(wpf, wpfk, C4 * C4);
    pack_conv_w<<<dim3((9 * C4 * C4) / 256), 256, 0, stream>>>(wcf, wcfk, C4 * C4);

    for (int b = 0; b < BATCH; ++b) {
        const float* xb = x + (size_t)b * CIN * NPIX;
        transpose_cast<<<dim3(NPIX / 64, CIN / 64), 256, 0, stream>>>(xb, NPIX, xbt, CIN);

        gemm_tn<1, 0, 2, 0><<<dim3(NPIX / 128, C16 / 128), 256, 0, stream>>>(
            wqpb, CIN, xbt, CIN, nullptr, 0, qpt, C16, C16, NPIX, CIN,
            bq_p, nullptr, nullptr, nullptr, nullptr);
        gemm_tn<2, 2, 2, 0><<<dim3(NPIX / 128, C4 / 128), 256, 0, stream>>>(
            wpcb, CIN, xbt, CIN, pcb, NPIX, pct, C4, C4, NPIX, CIN,
            bpc, spc, tpc, nullptr, nullptr);
        gemm_tn<1, 0, 2, 0><<<dim3(NPIX / 128, C16 / 128), 256, 0, stream>>>(
            wkb, C4, pct, C4, nullptr, 0, kpt, C16, C16, NPIX, C4,
            bk, nullptr, nullptr, nullptr, nullptr);
        gemm_tn<1, 2, 0, 0><<<dim3(NPIX / 128, C4 / 128), 256, 0, stream>>>(
            wvb, CIN, xbt, CIN, vpb, NPIX, nullptr, 0, C4, NPIX, CIN,
            bv, nullptr, nullptr, nullptr, nullptr);
        bf16_to_fp8<<<dim3((C4 * NPIX) / 256), 256, 0, stream>>>(vpb, vp8, C4 * NPIX);
        gemm_tn<0, 1, 0, 0><<<dim3(NPIX / 128, NPIX / 128), 256, 0, stream>>>(
            qpt, C16, kpt, C16, Smat, NPIX, nullptr, 0, NPIX, NPIX, C16,
            nullptr, nullptr, nullptr, nullptr, nullptr);
        softmax_row_fp8<<<NPIX, 256, 0, stream>>>(Smat, Pb8, rsum, NPIX);
        gemm_fp8_pv<<<dim3(C4 / 128, NPIX / 128), 256, 0, stream>>>(
            Pb8, vp8, rsum, Of, C4);
        gemm_tn<3, 0, 2, 1><<<dim3(NPIX / 128, C4 / 128), 256, 0, stream>>>(
            wrpb, CIN, xbt, CIN, nullptr, 0, ppad, C4, C4, NPIX, CIN,
            brp, srp, trp, Of, gam_p);
        conv3x3_wmma<<<dim3(HH, C4 / 64), 256, 0, stream>>>(
            ppad, wpfk, bpf, spf, tpf, out_p + (size_t)b * C4 * NPIX, popt);

        gemm_tn<1, 2, 0, 0><<<dim3(NPIX / 128, C4 / 128), 256, 0, stream>>>(
            wqcb, CIN, xbt, CIN, qcb, NPIX, nullptr, 0, C4, NPIX, CIN,
            bq_c, nullptr, nullptr, nullptr, nullptr);
        gemm_tn<0, 1, 0, 0><<<dim3(C4 / 128, C4 / 128), 256, 0, stream>>>(
            qcb, NPIX, pcb, NPIX, Emat, C4, nullptr, 0, C4, C4, NPIX,
            nullptr, nullptr, nullptr, nullptr, nullptr);
        softmax_chan<<<C4, 256, 0, stream>>>(Emat, acb, C4);
        gemm_tn<0, 0, 1, 0><<<dim3(NPIX / 128, C4 / 128), 256, 0, stream>>>(
            acb, C4, popt, C4, nullptr, 0, Of, C4, C4, NPIX, C4,
            nullptr, nullptr, nullptr, nullptr, nullptr);
        gemm_tn<3, 0, 2, 1><<<dim3(NPIX / 128, C4 / 128), 256, 0, stream>>>(
            wrcb, CIN, xbt, CIN, nullptr, 0, ppad, C4, C4, NPIX, CIN,
            brc, src_, trc, Of, gam_c);
        conv3x3_wmma<<<dim3(HH, C4 / 64), 256, 0, stream>>>(
            ppad, wcfk, bcf, scf, tcf, out_c + (size_t)b * C4 * NPIX, nullptr);
    }
}
